// NoiseGradationAdaptiveProposal_4243427688523
// MI455X (gfx1250) — matrix operand order repaired and checked by reading, not compiled, not run on hardware
//
#include <hip/hip_runtime.h>
#include <cstdint>

// ---------------------------------------------------------------------------
// NoiseGradationAdaptiveProposal — fused GMM log-likelihood for MI455X (gfx1250)
//   logits[g,s,x] = -0.5*(A[g,s] - 2*B[g,s,x] + C[g,x]) + const[g]
//   out[s] = logsumexp_{g,x} logits - log(X*G)
// B is a [S,F]x[F,X] GEMM per g on bf16 WMMA (hi/lo bf16x3 split, fp32 acc).
// Operands are staged in WMMA-fragment-order tiles (1KB per 16x32 tile,
// lane-major) so every fragment load is a contiguous, fully-coalesced
// 32B-per-lane read. Logsumexp is fused into the GEMM epilogue per tile.
// ---------------------------------------------------------------------------

typedef __attribute__((ext_vector_type(16))) __bf16 v16bf;
typedef __attribute__((ext_vector_type(8)))  float  v8f;

constexpr int S  = 2048;
constexpr int Xn = 2048;
constexpr int Gn = 2;
constexpr int Fn = 3072;
constexpr int BM = 128, BN = 128;
constexpr int ST = S  / BM;   // 16 s-tiles
constexpr int XT = Xn / BN;   // 16 x-tiles
constexpr int KT = Fn / 32;   // 96 k-tiles
constexpr int TILE = 512;     // elements per 16x32 bf16 fragment tile (1KB)

// ---- bf16 helpers (round-to-nearest-even split) ---------------------------
__device__ inline unsigned short f2bf(float x) {
  unsigned u = __float_as_uint(x);
  unsigned r = (u + 0x7FFFu + ((u >> 16) & 1u)) >> 16;
  return (unsigned short)r;
}
__device__ inline float bf2f(unsigned short h) {
  return __uint_as_float(((unsigned)h) << 16);
}

// ---- Pass 1a: P_g[s,f] = samples[s,f] / std[g,f]^2, hi/lo bf16, stored in
// A-fragment order: [g][S/16][F/32][lane(32)][elem(16)] -------------------
__global__ __launch_bounds__(256) void prep_p_kernel(const float* __restrict__ smp,
                                                     const float* __restrict__ stdv,
                                                     unsigned short* __restrict__ phi,
                                                     unsigned short* __restrict__ plo) {
  size_t i = (size_t)blockIdx.x * 256 + threadIdx.x;   // over G*S*(F/8)
  if (i >= (size_t)Gn * S * (Fn / 8)) return;
  int j8 = (int)(i % (Fn / 8));
  size_t gm = i / (Fn / 8);
  int g = (int)(gm / S);
  int m = (int)(gm % S);
  int k = j8 * 8;

  union { unsigned short s[8]; uint4 q; } hb, lb;
#pragma unroll
  for (int t = 0; t < 8; ++t) {
    float sd = stdv[g * Fn + k + t];
    float w  = 1.0f / (sd * sd);
    float p  = smp[(size_t)m * Fn + k + t] * w;
    unsigned short h = f2bf(p);
    hb.s[t] = h;
    lb.s[t] = f2bf(p - bf2f(h));
  }
  // A-fragment order (16x32 bf16): lane<16 -> M=lane, elems0-15=K0-15;
  // lane>=16 -> K16-31.
  int rt = m >> 4, mr = m & 15;
  int kt = k >> 5, ko = k & 31;
  int half = (ko >> 4) & 1;
  int eo   = (ko & 8) ? 8 : 0;
  int lane = half * 16 + mr;
  size_t off = (((size_t)g * (S / 16) + rt) * KT + kt) * TILE + lane * 16 + eo;
  *(uint4*)(phi + off) = hb.q;
  *(uint4*)(plo + off) = lb.q;
}

// ---- Pass 1b: x split hi/lo, stored in B-fragment order:
// [X/16][F/32][lane(32)][elem(16)]; lane<16 -> N=lane, K0-15; lane>=16 K16-31.
__global__ __launch_bounds__(256) void prep_x_kernel(const float* __restrict__ xin,
                                                     unsigned short* __restrict__ xhi,
                                                     unsigned short* __restrict__ xlo) {
  size_t i = (size_t)blockIdx.x * 256 + threadIdx.x;   // over X*(F/16)
  if (i >= (size_t)Xn * (Fn / 16)) return;
  int run = (int)(i % (Fn / 16));
  int n   = (int)(i / (Fn / 16));
  int k   = run * 16;

  union { unsigned short s[16]; uint4 q[2]; } hb, lb;
#pragma unroll
  for (int t = 0; t < 16; ++t) {
    float v = xin[(size_t)n * Fn + k + t];
    unsigned short h = f2bf(v);
    hb.s[t] = h;
    lb.s[t] = f2bf(v - bf2f(h));
  }
  int nt = n >> 4, nr = n & 15;
  int kt = k >> 5;
  int half = (k >> 4) & 1;
  int lane = half * 16 + nr;
  size_t off = ((size_t)nt * KT + kt) * TILE + lane * 16;
  *(uint4*)(xhi + off)     = hb.q[0];
  *(uint4*)(xhi + off + 8) = hb.q[1];
  *(uint4*)(xlo + off)     = lb.q[0];
  *(uint4*)(xlo + off + 8) = lb.q[1];
}

// ---- Row norms in fp32: out[g*rows + r] = sum_f mat[r,f]^2 / std[g,f]^2 ---
__global__ __launch_bounds__(256) void rownorm_kernel(const float* __restrict__ mat,
                                                      const float* __restrict__ stdv,
                                                      float* __restrict__ out, int rows) {
  int b = blockIdx.x;
  int g = b / rows, r = b % rows;
  __shared__ float red[256];
  float sum = 0.0f;
  for (int f = threadIdx.x; f < Fn; f += 256) {
    float v  = mat[(size_t)r * Fn + f];
    float sd = stdv[g * Fn + f];
    sum += (v * v) / (sd * sd);
  }
  red[threadIdx.x] = sum;
  __syncthreads();
  for (int sN = 128; sN > 0; sN >>= 1) {
    if (threadIdx.x < sN) red[threadIdx.x] += red[threadIdx.x + sN];
    __syncthreads();
  }
  if (threadIdx.x == 0) out[b] = red[0];
}

// ---- const[g] = -0.5*F*log(2pi) - sum_f log(std[g,f]) ---------------------
__global__ __launch_bounds__(256) void const_kernel(const float* __restrict__ stdv,
                                                    float* __restrict__ cst) {
  int g = blockIdx.x;
  __shared__ float red[256];
  float sum = 0.0f;
  for (int f = threadIdx.x; f < Fn; f += 256) sum += __logf(stdv[g * Fn + f]);
  red[threadIdx.x] = sum;
  __syncthreads();
  for (int sN = 128; sN > 0; sN >>= 1) {
    if (threadIdx.x < sN) red[threadIdx.x] += red[threadIdx.x + sN];
    __syncthreads();
  }
  if (threadIdx.x == 0) cst[g] = -0.5f * (float)Fn * 1.8378770664093453f - red[0];
}

// ---- Fragment load: 32 contiguous bytes per lane from a 1KB tile ----------
__device__ inline v16bf load_frag(const unsigned short* p) {
  union { uint4 q[2]; v16bf v; } u;
  u.q[0] = *(const uint4*)(p);
  u.q[1] = *(const uint4*)(p + 8);
  return u.v;
}

// ---- Main fused kernel: 128x128 tile GEMM (bf16x3) + online logsumexp -----
// grid = Gn*ST*XT blocks, 256 threads (8 waves); wave w owns rows
// [w*16, w*16+16) of the tile across all 128 columns.
__global__ __launch_bounds__(256) void gemm_lse_kernel(
    const unsigned short* __restrict__ Phi, const unsigned short* __restrict__ Plo,
    const unsigned short* __restrict__ Xhi, const unsigned short* __restrict__ Xlo,
    const float* __restrict__ Arow, const float* __restrict__ Crow,
    float* __restrict__ PM, float* __restrict__ PL) {
  int b  = blockIdx.x;
  int xt = b & (XT - 1);
  int st = (b >> 4) & (ST - 1);
  int g  = b >> 8;
  int tid  = threadIdx.x;
  int wid  = tid >> 5;
  int lane = tid & 31;
  int half = lane >> 4;
  int nr   = lane & 15;

  // A-side base (fragment-order tiles), one 16-row subtile per wave
  int rt = st * 8 + wid;
  const unsigned short* pAhi =
      Phi + (((size_t)g * (S / 16) + rt) * KT) * TILE + lane * 16;
  const unsigned short* pAlo =
      Plo + (((size_t)g * (S / 16) + rt) * KT) * TILE + lane * 16;

  // B-side bases, one per 16-col subtile
  const unsigned short* pBhi[8];
  const unsigned short* pBlo[8];
#pragma unroll
  for (int ct = 0; ct < 8; ++ct) {
    size_t base = ((size_t)(xt * 8 + ct) * KT) * TILE + lane * 16;
    pBhi[ct] = Xhi + base;
    pBlo[ct] = Xlo + base;
  }

  v8f acc[8] = {};

  for (int kt = 0; kt < KT; ++kt) {
    size_t ko = (size_t)kt * TILE;
    if (kt + 1 < KT) {
      __builtin_prefetch(pAhi + ko + TILE, 0, 0);
      __builtin_prefetch(pAlo + ko + TILE, 0, 0);
    }
    v16bf ah = load_frag(pAhi + ko);
    v16bf al = load_frag(pAlo + ko);
#pragma unroll
    for (int ct = 0; ct < 8; ++ct) {
      v16bf bh = load_frag(pBhi[ct] + ko);
      v16bf bl = load_frag(pBlo[ct] + ko);
      // bf16x3: hi*hi + hi*lo + lo*hi, fp32 accumulate
      acc[ct] = __builtin_amdgcn_wmma_f32_16x16x32_bf16(false, ah, false, bh,
                                                        (short)0, acc[ct], false, false);
      acc[ct] = __builtin_amdgcn_wmma_f32_16x16x32_bf16(false, ah, false, bl,
                                                        (short)0, acc[ct], false, false);
      acc[ct] = __builtin_amdgcn_wmma_f32_16x16x32_bf16(false, al, false, bh,
                                                        (short)0, acc[ct], false, false);
    }
  }

  // Epilogue: logits = B - 0.5*(A_row + C_col); per-row max + sumexp.
  // C/D layout: VGPR r, lanes 0-15 -> (M=r, N=lane); lanes 16-31 -> (M=8+r).
  float rowA[8], colC[8];
#pragma unroll
  for (int r = 0; r < 8; ++r)
    rowA[r] = Arow[g * S + st * BM + wid * 16 + half * 8 + r];
#pragma unroll
  for (int ct = 0; ct < 8; ++ct)
    colC[ct] = Crow[g * Xn + xt * BN + ct * 16 + nr];

#pragma unroll
  for (int ct = 0; ct < 8; ++ct)
#pragma unroll
    for (int r = 0; r < 8; ++r)
      acc[ct][r] -= 0.5f * (rowA[r] + colC[ct]);

#pragma unroll
  for (int r = 0; r < 8; ++r) {
    float m = acc[0][r];
#pragma unroll
    for (int ct = 1; ct < 8; ++ct) m = fmaxf(m, acc[ct][r]);
    // butterfly across the 16 lanes holding this row's columns (wave32)
    m = fmaxf(m, __shfl_xor(m, 1, 32));
    m = fmaxf(m, __shfl_xor(m, 2, 32));
    m = fmaxf(m, __shfl_xor(m, 4, 32));
    m = fmaxf(m, __shfl_xor(m, 8, 32));
    float l = 0.0f;
#pragma unroll
    for (int ct = 0; ct < 8; ++ct) l += __expf(acc[ct][r] - m);
    l += __shfl_xor(l, 1, 32);
    l += __shfl_xor(l, 2, 32);
    l += __shfl_xor(l, 4, 32);
    l += __shfl_xor(l, 8, 32);
    if (nr == r) {  // lanes (r) and (16+r) write rows wid*16+r and wid*16+8+r
      int row = wid * 16 + half * 8 + r;
      size_t o = (((size_t)g * ST + st) * XT + xt) * BM + row;
      PM[o] = m;
      PL[o] = l;
    }
  }
}

// ---- Final merge over (g, x-chunks): lp[s] ---------------------------------
__global__ __launch_bounds__(256) void finalize_kernel(const float* __restrict__ PM,
                                                       const float* __restrict__ PL,
                                                       const float* __restrict__ cst,
                                                       float* __restrict__ out) {
  int s = blockIdx.x * 256 + threadIdx.x;
  if (s >= S) return;
  int st = s >> 7, row = s & 127;
  float M = -3.4e38f, L = 0.0f;
  for (int g = 0; g < Gn; ++g) {
    float cg = cst[g];
    for (int xt = 0; xt < XT; ++xt) {
      size_t o = (((size_t)g * ST + st) * XT + xt) * BM + row;
      float m = PM[o] + cg;
      float l = PL[o];
      if (m > M) { L = L * __expf(M - m) + l; M = m; }
      else       { L = L + l * __expf(m - M); }
    }
  }
  out[s] = M + __logf(L) - 8.317766166719343f;  // log(X*G) = log(4096)
}

// ---------------------------------------------------------------------------
extern "C" void kernel_launch(void* const* d_in, const int* in_sizes, int n_in,
                              void* d_out, int out_size, void* d_ws, size_t ws_size,
                              hipStream_t stream) {
  (void)in_sizes; (void)n_in; (void)out_size; (void)ws_size;
  const float* smp  = (const float*)d_in[0];  // [S, F]
  const float* xin  = (const float*)d_in[1];  // [X, F]
  const float* stdv = (const float*)d_in[2];  // [G, F]
  float* out = (float*)d_out;                 // [S]

  char* w = (char*)d_ws;
  auto align256 = [](size_t v) { return (v + 255) & ~(size_t)255; };
  size_t off = 0;
  unsigned short* Phi = (unsigned short*)(w + off); off = align256(off + (size_t)Gn * S * Fn * 2);
  unsigned short* Plo = (unsigned short*)(w + off); off = align256(off + (size_t)Gn * S * Fn * 2);
  unsigned short* Xhi = (unsigned short*)(w + off); off = align256(off + (size_t)Xn * Fn * 2);
  unsigned short* Xlo = (unsigned short*)(w + off); off = align256(off + (size_t)Xn * Fn * 2);
  float* Arow = (float*)(w + off); off = align256(off + (size_t)Gn * S * 4);
  float* Crow = (float*)(w + off); off = align256(off + (size_t)Gn * Xn * 4);
  float* Cst  = (float*)(w + off); off = align256(off + 256);
  float* PM   = (float*)(w + off); off = align256(off + (size_t)Gn * ST * XT * BM * 4);
  float* PL   = (float*)(w + off); off = align256(off + (size_t)Gn * ST * XT * BM * 4);

  {
    size_t n = (size_t)Gn * S * (Fn / 8);
    prep_p_kernel<<<(unsigned)((n + 255) / 256), 256, 0, stream>>>(smp, stdv, Phi, Plo);
  }
  {
    size_t n = (size_t)Xn * (Fn / 16);
    prep_x_kernel<<<(unsigned)((n + 255) / 256), 256, 0, stream>>>(xin, Xhi, Xlo);
  }
  rownorm_kernel<<<Gn * S,  256, 0, stream>>>(smp, stdv, Arow, S);
  rownorm_kernel<<<Gn * Xn, 256, 0, stream>>>(xin, stdv, Crow, Xn);
  const_kernel<<<Gn, 256, 0, stream>>>(stdv, Cst);
  gemm_lse_kernel<<<Gn * ST * XT, 256, 0, stream>>>(Phi, Plo, Xhi, Xlo, Arow, Crow, PM, PL);
  finalize_kernel<<<(S + 255) / 256, 256, 0, stream>>>(PM, PL, Cst, out);
}
